// RAM_41223096107152
// MI455X (gfx1250) — matrix operand order repaired and checked by reading, not compiled, not run on hardware
//
#include <hip/hip_runtime.h>
#include <hip/hip_bf16.h>

// Problem constants
#define B_    256
#define S_    128
#define AS_   5
#define E_    300
#define H_    300
#define P_    3
#define HOPS_ 3
#define KP    320   // K (=E=H=300) padded to multiple of 32
#define NPG   304   // per-gate N (=300) padded to multiple of 16
#define NT    19    // N tiles per gate (304/16)
#define GATES 4

typedef __bf16 bf8   __attribute__((ext_vector_type(8)));
typedef __bf16 v16bf __attribute__((ext_vector_type(16)));
typedef float  v8f   __attribute__((ext_vector_type(8)));

__device__ inline __bf16 f2bf(float f) {
    unsigned u = __builtin_bit_cast(unsigned, f);
    u += 0x7FFFu + ((u >> 16) & 1u);   // round-to-nearest-even
    unsigned short s = (unsigned short)(u >> 16);
    return __builtin_bit_cast(__bf16, s);
}

__device__ inline float sigm(float x) { return 1.0f / (1.0f + expf(-x)); }

// Build a v16bf from two 8-element (16B) chunks -> two global_load_b128
__device__ inline v16bf ld16_split(const __bf16* lo, const __bf16* hi) {
    bf8 a = *(const bf8*)lo;
    bf8 b = *(const bf8*)hi;
    v16bf r;
#pragma unroll
    for (int i = 0; i < 8; ++i) { r[i] = a[i]; r[i + 8] = b[i]; }
    return r;
}

__device__ inline v8f wmma_bf16(v16bf a, v16bf b, v8f c) {
    // (neg_a, A, neg_b, B, c_mod, C, reuse_a, reuse_b)
    return __builtin_amdgcn_wmma_f32_16x16x32_bf16(false, a, false, b, (short)0, c, false, false);
}

// ---------------- small prep kernels ----------------

__global__ void lengths_kernel(const int* __restrict__ traw, const int* __restrict__ tasp,
                               const int* __restrict__ tleft,
                               int* lmem, int* lasp, int* lleft) {
    int b = threadIdx.x;
    if (b >= B_) return;
    int m = 0, l = 0, a = 0;
    for (int s = 0; s < S_; ++s) { m += (traw[b * S_ + s] != 0); l += (tleft[b * S_ + s] != 0); }
    for (int s = 0; s < AS_; ++s) a += (tasp[b * AS_ + s] != 0);
    lmem[b] = m; lleft[b] = l; lasp[b] = a;
}

// src: [1200][300] f32 (gate-major rows) -> dst: [4][NPG][KP] bf16, zero padded
__global__ void conv_w_kernel(const float* __restrict__ src, __bf16* __restrict__ dst) {
    int i = blockIdx.x * blockDim.x + threadIdx.x;
    if (i >= GATES * NPG * KP) return;
    int k = i % KP;
    int n = (i / KP) % NPG;
    int g = i / (KP * NPG);
    float v = (n < H_ && k < H_) ? src[(g * H_ + n) * H_ + k] : 0.0f;
    dst[i] = f2bf(v);
}

// combined bias (b_ih + b_hh) padded to [4][NPG]
__global__ void bias_kernel(const float* __restrict__ bih, const float* __restrict__ bhh,
                            float* __restrict__ dst) {
    int i = blockIdx.x * blockDim.x + threadIdx.x;
    if (i >= GATES * NPG) return;
    int n = i % NPG;
    int g = i / NPG;
    dst[i] = (n < H_) ? (bih[g * H_ + n] + bhh[g * H_ + n]) : 0.0f;
}

// embedding gather -> X bf16 [B*S][KP] (K padded with zeros)
__global__ void embed_x_kernel(const int* __restrict__ traw, const float* __restrict__ emb,
                               __bf16* __restrict__ X) {
    long i = (long)blockIdx.x * blockDim.x + threadIdx.x;
    if (i >= (long)B_ * S_ * KP) return;
    int  k   = (int)(i % KP);
    long row = i / KP;
    int  tok = traw[row];
    X[i] = (k < E_) ? f2bf(emb[(long)tok * E_ + k]) : f2bf(0.0f);
}

// ---------------- fused LSTM step (WMMA GEMM + cell) ----------------
// gates(256x1216) = x_t(256x320) @ Wih^T + h(256x320) @ Whh^T  (bf16 in, f32 acc)
// One wave owns one 16x16 tile of each of the 4 gates, then applies the cell.
__global__ __launch_bounds__(128) void lstm_step_kernel(
    const __bf16* __restrict__ X,      // [B*S][KP]
    const __bf16* __restrict__ Wih,    // [4][NPG][KP]
    const __bf16* __restrict__ Whh,    // [4][NPG][KP]
    const float*  __restrict__ bias,   // [4][NPG]
    const int*    __restrict__ lmem,
    const __bf16* __restrict__ h_in,   // [B][KP]   (double-buffered)
    __bf16*       __restrict__ h_out,  // [B][KP]
    float*        __restrict__ hF,     // [B][NPG]  f32 h (for mask-keep)
    float*        __restrict__ cF,     // [B][NPG]
    float*        __restrict__ outseq, // [B][S][H]
    int t, int dir) {
    int wave = blockIdx.x * 4 + (threadIdx.x >> 5);   // 76 blocks * 4 waves = 304 = 16*19
    int lane = threadIdx.x & 31;
    int mt = wave / NT, nt = wave % NT;
    int m_base = mt * 16, n_base = nt * 16;
    int lh = lane >> 4, lm = lane & 15;

    // A-operand row for this lane (per documented 16-bit A layout: M = lane&15)
    int arow = m_base + lm;
    int L = lmem[arow];
    int xs = t;
    if (dir) { int r = L - 1 - t; xs = r < 0 ? 0 : (r > S_ - 1 ? S_ - 1 : r); }
    const __bf16* xrow = X + ((long)(arow * S_ + xs)) * KP;
    const __bf16* hrow = h_in + (long)arow * KP;
    int nw = n_base + lm;   // B-operand row (weight row), also C-layout N

    v8f acc[GATES] = { v8f{}, v8f{}, v8f{}, v8f{} };

    for (int kb = 0; kb < KP / 32; ++kb) {
        // A layout: lanes 0-15 hold K 0..7 / 16..23, lanes 16-31 hold K 8..15 / 24..31
        int alo = kb * 32 + lh * 8;
        v16bf ax = ld16_split(xrow + alo, xrow + alo + 16);
        v16bf ah = ld16_split(hrow + alo, hrow + alo + 16);
        // B layout: lanes 0-15 hold K 0..7 / 16..23, lanes 16-31 hold K 8..15 / 24..31
        int boff = kb * 32 + lh * 8;
#pragma unroll
        for (int g = 0; g < GATES; ++g) {
            const __bf16* wi = Wih + ((long)(g * NPG + nw)) * KP + boff;
            const __bf16* wh = Whh + ((long)(g * NPG + nw)) * KP + boff;
            v16bf bi = ld16_split(wi, wi + 16);
            acc[g] = wmma_bf16(ax, bi, acc[g]);
            v16bf bh = ld16_split(wh, wh + 16);
            acc[g] = wmma_bf16(ah, bh, acc[g]);
        }
    }

    // C/D layout: VGPR r, lanes 0-15 -> M=r, lanes 16-31 -> M=8+r ; N = lane&15
#pragma unroll
    for (int r = 0; r < 8; ++r) {
        int m = m_base + lh * 8 + r;
        int n = nw;
        float iv = acc[0][r] + bias[0 * NPG + n];
        float fv = acc[1][r] + bias[1 * NPG + n];
        float gv = acc[2][r] + bias[2 * NPG + n];
        float ov = acc[3][r] + bias[3 * NPG + n];
        long idx = (long)m * NPG + n;
        float c_old = cF[idx];
        float c_new = sigm(fv) * c_old + sigm(iv) * tanhf(gv);
        float h_new = sigm(ov) * tanhf(c_new);
        bool act = t < lmem[m];
        float h2 = act ? h_new : hF[idx];
        cF[idx] = act ? c_new : c_old;
        hF[idx] = h2;
        h_out[(long)m * KP + n] = f2bf(h2);
        if (n < H_) outseq[((long)m * S_ + t) * H_ + n] = act ? h_new : 0.0f;
    }
}

// ---------------- attention (scores + softmax + i_t), computed once ----------------
// et/aspect/att_b terms are constant over the sequence axis -> cancel in softmax.
__global__ __launch_bounds__(128) void attention_kernel(
    const float* __restrict__ outF, const float* __restrict__ outB,
    const float* __restrict__ attw,
    const int* __restrict__ lmem, const int* __restrict__ lasp, const int* __restrict__ lleft,
    float* __restrict__ itv /* [B][601] */) {
    __shared__ float s_aw[S_];
    __shared__ float s_red[S_];
    __shared__ float s_udot;
    int b = blockIdx.x, tid = threadIdx.x;
    int L = lmem[b], A = lasp[b], LL = lleft[b];
    int s = tid;

    // location weighting
    float u, w;
    {
        int a0 = LL, a1 = LL + A;
        float uu, ll;
        if (s < a0)      { uu = (float)(s - a0); ll = (float)(a0 - s); }
        else if (s < a1) { uu = 0.0f; ll = 0.0f; }
        else             { uu = (float)(s - a1 + 1); ll = uu; }
        bool valid = s < L;
        u = valid ? uu : 0.0f;
        w = valid ? (1.0f - ll / (float)L) : 1.0f;
    }

    // score
    float sc = 0.0f;
    const float* fr = outF + ((long)b * S_ + s) * H_;
    for (int e = 0; e < H_; ++e) sc += fr[e] * attw[e];
    float scb = 0.0f;
    if (s < L) {
        int rev = L - 1 - s;
        const float* br = outB + ((long)b * S_ + rev) * H_;
        for (int e = 0; e < H_; ++e) scb += br[e] * attw[H_ + e];
    }
    sc = (sc + scb) * w + u * attw[2 * H_];

    // softmax over s
    s_red[tid] = sc; __syncthreads();
    for (int off = 64; off > 0; off >>= 1) { if (tid < off) s_red[tid] = fmaxf(s_red[tid], s_red[tid + off]); __syncthreads(); }
    float mx = s_red[0]; __syncthreads();
    float ex = expf(sc - mx);
    s_red[tid] = ex; __syncthreads();
    for (int off = 64; off > 0; off >>= 1) { if (tid < off) s_red[tid] += s_red[tid + off]; __syncthreads(); }
    float inv = 1.0f / s_red[0]; __syncthreads();
    float alpha = ex * inv;
    s_aw[tid] = alpha * w;               // alpha * location-weight
    s_red[tid] = alpha * u; __syncthreads();
    for (int off = 64; off > 0; off >>= 1) { if (tid < off) s_red[tid] += s_red[tid + off]; __syncthreads(); }
    if (tid == 0) s_udot = s_red[0];
    __syncthreads();

    // i_t = sum_s alpha * (location-weighted memory, u)
    for (int d = tid; d <= 2 * H_; d += 128) {
        float acc2 = 0.0f;
        if (d < H_) {
            for (int s2 = 0; s2 < S_; ++s2)
                acc2 += s_aw[s2] * outF[((long)b * S_ + s2) * H_ + d];
        } else if (d < 2 * H_) {
            int e = d - H_;
            for (int s2 = 0; s2 < L; ++s2) {
                int rev = L - 1 - s2;
                acc2 += s_aw[s2] * outB[((long)b * S_ + rev) * H_ + e];
            }
        } else {
            acc2 = s_udot;
        }
        itv[(long)b * (2 * H_ + 1) + d] = acc2;
    }
}

// gi = i_t @ gru_w_ih^T + b_ih  (hop-invariant, computed once)
__global__ void gi_kernel(const float* __restrict__ itv, const float* __restrict__ gwih,
                          const float* __restrict__ gbih, float* __restrict__ gi) {
    int i = blockIdx.x * blockDim.x + threadIdx.x;
    if (i >= B_ * 3 * E_) return;
    int j = i % (3 * E_);
    int b = i / (3 * E_);
    const int D = 2 * H_ + 1;
    float a = gbih[j];
    const float* it = itv + (long)b * D;
    const float* wr = gwih + (long)j * D;
    for (int d = 0; d < D; ++d) a += it[d] * wr[d];
    gi[i] = a;
}

// one GRU hop: gh = et @ gru_w_hh^T + b_hh ; et = (1-z)*n + z*et
__global__ __launch_bounds__(256) void hop_kernel(const float* __restrict__ gi,
                                                  const float* __restrict__ gwhh,
                                                  const float* __restrict__ gbhh,
                                                  float* __restrict__ et) {
    __shared__ float s_et[E_];
    __shared__ float s_gh[3 * E_];
    int b = blockIdx.x, tid = threadIdx.x;
    for (int e = tid; e < E_; e += 256) s_et[e] = et[(long)b * E_ + e];
    __syncthreads();
    for (int j = tid; j < 3 * E_; j += 256) {
        float a = gbhh[j];
        const float* wr = gwhh + (long)j * E_;
        for (int d = 0; d < E_; ++d) a += s_et[d] * wr[d];
        s_gh[j] = a;
    }
    __syncthreads();
    const float* gib = gi + (long)b * 3 * E_;
    for (int e = tid; e < E_; e += 256) {
        float r = sigm(gib[e] + s_gh[e]);
        float z = sigm(gib[E_ + e] + s_gh[E_ + e]);
        float n = tanhf(gib[2 * E_ + e] + r * s_gh[2 * E_ + e]);
        et[(long)b * E_ + e] = (1.0f - z) * n + z * s_et[e];
    }
}

__global__ __launch_bounds__(128) void dense_kernel(const float* __restrict__ et,
                                                    const float* __restrict__ dw,
                                                    const float* __restrict__ db,
                                                    float* __restrict__ out) {
    __shared__ float red[128];
    int b = blockIdx.x, tid = threadIdx.x;
    for (int p = 0; p < P_; ++p) {
        float a = 0.0f;
        for (int e = tid; e < E_; e += 128) a += et[(long)b * E_ + e] * dw[p * E_ + e];
        red[tid] = a; __syncthreads();
        for (int off = 64; off > 0; off >>= 1) { if (tid < off) red[tid] += red[tid + off]; __syncthreads(); }
        if (tid == 0) out[b * P_ + p] = red[0] + db[p];
        __syncthreads();
    }
}

// ---------------- host orchestration ----------------

extern "C" void kernel_launch(void* const* d_in, const int* in_sizes, int n_in,
                              void* d_out, int out_size, void* d_ws, size_t ws_size,
                              hipStream_t stream) {
    (void)in_sizes; (void)n_in; (void)out_size; (void)ws_size;

    const int*   traw  = (const int*)d_in[0];
    const int*   tasp  = (const int*)d_in[1];
    const int*   tleft = (const int*)d_in[2];
    const float* emb   = (const float*)d_in[3];
    const float* wih_f = (const float*)d_in[4];
    const float* whh_f = (const float*)d_in[5];
    const float* bih_f = (const float*)d_in[6];
    const float* bhh_f = (const float*)d_in[7];
    const float* wih_b = (const float*)d_in[8];
    const float* whh_b = (const float*)d_in[9];
    const float* bih_b = (const float*)d_in[10];
    const float* bhh_b = (const float*)d_in[11];
    const float* attw  = (const float*)d_in[12];
    // d_in[13] (att_b) cancels in softmax; aspect embedding likewise unused.
    const float* gwih  = (const float*)d_in[14];
    const float* gwhh  = (const float*)d_in[15];
    const float* gbih  = (const float*)d_in[16];
    const float* gbhh  = (const float*)d_in[17];
    const float* dw    = (const float*)d_in[18];
    const float* db    = (const float*)d_in[19];
    float* out = (float*)d_out;

    char* base = (char*)d_ws;
    size_t off = 0;
    auto take = [&](size_t bytes) -> char* {
        char* p = base + off;
        off += (bytes + 255) & ~(size_t)255;
        return p;
    };

    int*    d_lmem  = (int*)take(B_ * 4);
    int*    d_lasp  = (int*)take(B_ * 4);
    int*    d_lleft = (int*)take(B_ * 4);
    __bf16* d_wihF  = (__bf16*)take((size_t)GATES * NPG * KP * 2);
    __bf16* d_whhF  = (__bf16*)take((size_t)GATES * NPG * KP * 2);
    __bf16* d_wihB  = (__bf16*)take((size_t)GATES * NPG * KP * 2);
    __bf16* d_whhB  = (__bf16*)take((size_t)GATES * NPG * KP * 2);
    float*  d_biasF = (float*)take((size_t)GATES * NPG * 4);
    float*  d_biasB = (float*)take((size_t)GATES * NPG * 4);
    __bf16* d_X     = (__bf16*)take((size_t)B_ * S_ * KP * 2);
    __bf16* d_hbfA  = (__bf16*)take((size_t)B_ * KP * 2);
    __bf16* d_hbfB  = (__bf16*)take((size_t)B_ * KP * 2);
    float*  d_hF    = (float*)take((size_t)B_ * NPG * 4);
    float*  d_cF    = (float*)take((size_t)B_ * NPG * 4);
    float*  d_outF  = (float*)take((size_t)B_ * S_ * H_ * 4);
    float*  d_outB  = (float*)take((size_t)B_ * S_ * H_ * 4);
    float*  d_itv   = (float*)take((size_t)B_ * (2 * H_ + 1) * 4);
    float*  d_gi    = (float*)take((size_t)B_ * 3 * E_ * 4);
    float*  d_et    = (float*)take((size_t)B_ * E_ * 4);

    // 1) sequence lengths
    lengths_kernel<<<1, 256, 0, stream>>>(traw, tasp, tleft, d_lmem, d_lasp, d_lleft);

    // 2) weight / bias prep (f32 -> padded bf16)
    {
        int n = GATES * NPG * KP, g = (n + 255) / 256;
        conv_w_kernel<<<g, 256, 0, stream>>>(wih_f, d_wihF);
        conv_w_kernel<<<g, 256, 0, stream>>>(whh_f, d_whhF);
        conv_w_kernel<<<g, 256, 0, stream>>>(wih_b, d_wihB);
        conv_w_kernel<<<g, 256, 0, stream>>>(whh_b, d_whhB);
        int nb = GATES * NPG, gb = (nb + 255) / 256;
        bias_kernel<<<gb, 256, 0, stream>>>(bih_f, bhh_f, d_biasF);
        bias_kernel<<<gb, 256, 0, stream>>>(bih_b, bhh_b, d_biasB);
    }

    // 3) embedding gather -> bf16 X
    {
        long n = (long)B_ * S_ * KP;
        embed_x_kernel<<<(int)((n + 255) / 256), 256, 0, stream>>>(traw, emb, d_X);
    }

    // 4) BiLSTM: forward then backward, fused WMMA GEMM + cell per step
    for (int dir = 0; dir < 2; ++dir) {
        hipMemsetAsync(d_hbfA, 0, (size_t)B_ * KP * 2, stream);
        hipMemsetAsync(d_hbfB, 0, (size_t)B_ * KP * 2, stream);
        hipMemsetAsync(d_hF,   0, (size_t)B_ * NPG * 4, stream);
        hipMemsetAsync(d_cF,   0, (size_t)B_ * NPG * 4, stream);
        const __bf16* Wih  = dir ? d_wihB  : d_wihF;
        const __bf16* Whh  = dir ? d_whhB  : d_whhF;
        const float*  bias = dir ? d_biasB : d_biasF;
        float*        oseq = dir ? d_outB  : d_outF;
        const __bf16* hin  = d_hbfA;
        __bf16*       hout = d_hbfB;
        for (int t = 0; t < S_; ++t) {
            lstm_step_kernel<<<76, 128, 0, stream>>>(d_X, Wih, Whh, bias, d_lmem,
                                                     hin, hout, d_hF, d_cF, oseq, t, dir);
            const __bf16* tmp = hin; hin = hout; hout = (__bf16*)tmp;
        }
    }

    // 5) attention (scores/softmax/i_t) — hop-invariant, computed once
    attention_kernel<<<B_, 128, 0, stream>>>(d_outF, d_outB, attw, d_lmem, d_lasp, d_lleft, d_itv);

    // 6) gi = i_t @ W_ih^T + b_ih (hop-invariant)
    {
        int n = B_ * 3 * E_;
        gi_kernel<<<(n + 255) / 256, 256, 0, stream>>>(d_itv, gwih, gbih, d_gi);
    }

    // 7) GRU hops
    hipMemsetAsync(d_et, 0, (size_t)B_ * E_ * 4, stream);
    for (int h = 0; h < HOPS_; ++h)
        hop_kernel<<<B_, 256, 0, stream>>>(d_gi, gwhh, gbhh, d_et);

    // 8) dense head
    dense_kernel<<<B_, 128, 0, stream>>>(d_et, dw, db, out);
}
